// xPosAttention_39006892982516
// MI455X (gfx1250) — matrix operand order repaired and checked by reading, not compiled, not run on hardware
//
#include <hip/hip_runtime.h>
#include <hip/hip_bf16.h>

typedef __attribute__((ext_vector_type(16))) __bf16 v16bf;
typedef __attribute__((ext_vector_type(8)))  float  v8f;

#define DIMM 512
#define NH   8
#define DH   64
#define SEQ  2048
#define BL   4                 // b*l
#define MTOT (BL*SEQ)          // 8192
#define LN10000 9.210340371976184f
#define ATT_SCALE 0.04419417382415922f   // 1/sqrt(512)

// ---------- helpers ----------
__device__ inline unsigned short f2bf(float f) {
  unsigned u = __float_as_uint(f);
  u += 0x7fffu + ((u >> 16) & 1u);          // round-to-nearest-even
  return (unsigned short)(u >> 16);
}

union FragBF { v16bf v; uint4 u[2]; };

// A-operand fragment (16xK slice): lane<16 -> K {0..7,16..23}, lane>=16 -> {8..15,24..31}
__device__ inline v16bf ldfragA(const unsigned short* rowptr, int kbase, int lane) {
  FragBF f;
  const unsigned short* p = rowptr + kbase + ((lane & 16) ? 8 : 0);
  f.u[0] = *(const uint4*)(p);
  f.u[1] = *(const uint4*)(p + 16);
  return f.v;
}
// B-operand fragment (Kx16 slice): lane<16 -> K {0..7,16..23}, lane>=16 -> {8..15,24..31}
__device__ inline v16bf ldfragB(const unsigned short* rowptr, int kbase, int lane) {
  FragBF f;
  const unsigned short* p = rowptr + kbase + ((lane & 16) ? 8 : 0);
  f.u[0] = *(const uint4*)(p);
  f.u[1] = *(const uint4*)(p + 16);
  return f.v;
}
__device__ inline v8f wmma_bf16(v16bf a, v16bf b, v8f c) {
  return __builtin_amdgcn_wmma_f32_16x16x32_bf16(false, a, false, b, (short)0, c, false, false);
}

// ---------- kernel 0: f32 -> bf16 conversion (vectorized x4) ----------
__global__ __launch_bounds__(256) void cvt_f32_bf16_v4(const float4* __restrict__ src,
                                                       uint2* __restrict__ dst, int n4) {
  int i = blockIdx.x * blockDim.x + threadIdx.x;
  if (i < n4) {
    float4 f = src[i];
    uint2 r;
    r.x = (unsigned)f2bf(f.x) | ((unsigned)f2bf(f.y) << 16);
    r.y = (unsigned)f2bf(f.z) | ((unsigned)f2bf(f.w) << 16);
    dst[i] = r;
  }
}

// ---------- kernel 1: QKV projections + bias + rotary (pipelined, fully unrolled) ----------
// p=0: Wq,bq -> rotary -> Q [head][s][dh]
// p=1: Wv,bv -> rotary -> K [head][s][dh]     (reference: k_mat = rotary(v_proj))
// p=2: Wk,bk -> (none) -> Vt [head][dh][s]    (reference: v_mat = k_proj), transposed for P@V
__global__ __launch_bounds__(256) void proj_kernel(
    const unsigned short* __restrict__ xb,
    const unsigned short* __restrict__ Wqb,
    const unsigned short* __restrict__ Wvb,
    const unsigned short* __restrict__ Wkb,
    const float* __restrict__ bq, const float* __restrict__ bv, const float* __restrict__ bk,
    unsigned short* __restrict__ Qws, unsigned short* __restrict__ Kws,
    unsigned short* __restrict__ Vt)
{
  const int lane = threadIdx.x & 31;
  const int w    = blockIdx.x * (blockDim.x >> 5) + (threadIdx.x >> 5);
  const int mt = w & 511;            // 512 M-tiles of 16
  const int ng = (w >> 9) & 7;       // 8 N-groups of 64
  const int p  = w >> 12;            // projection id
  const int m0 = mt * 16, e0 = ng * 64;
  const int colL = lane & 15, lh = lane >> 4;

  const unsigned short* Wb  = (p == 0) ? Wqb : (p == 1) ? Wvb : Wkb;
  const float*          bias= (p == 0) ? bq  : (p == 1) ? bv  : bk;

  const unsigned short* arow = xb + (size_t)(m0 + colL) * DIMM;
  const unsigned short* wrow[4];
#pragma unroll
  for (int t = 0; t < 4; ++t) wrow[t] = Wb + (size_t)(e0 + t * 16 + colL) * DIMM;

  // software pipeline: fragments for step k+32 in flight while step k computes.
  // Fully unrolled so ping-pong buffers get distinct registers (no v_mov shuffle).
  v16bf a = ldfragA(arow, 0, lane);
  v16bf b[4];
#pragma unroll
  for (int t = 0; t < 4; ++t) b[t] = ldfragB(wrow[t], 0, lane);

  v8f acc[4] = {};
#pragma unroll
  for (int k = 0; k < DIMM; k += 32) {
    const int kn = (k + 32 < DIMM) ? k + 32 : k;   // last iter: harmless reload
    v16bf an = ldfragA(arow, kn, lane);
    v16bf bn[4];
#pragma unroll
    for (int t = 0; t < 4; ++t) bn[t] = ldfragB(wrow[t], kn, lane);
#pragma unroll
    for (int t = 0; t < 4; ++t) acc[t] = wmma_bf16(a, b[t], acc[t]);
    a = an;
#pragma unroll
    for (int t = 0; t < 4; ++t) b[t] = bn[t];
  }

#pragma unroll
  for (int t = 0; t < 4; ++t) {
    const int e = e0 + t * 16 + colL;
    const float bval = bias[e];
    const int h = e >> 6, f = e & 63;
    const float invf = __expf(-((float)(f & ~1) * (1.0f / 64.0f)) * LN10000);
#pragma unroll
    for (int j = 0; j < 8; ++j) {
      const int m = m0 + j + lh * 8;
      const int bl = m >> 11, spos = m & 2047;
      float val = acc[t][j] + bval;
      if (p < 2) {                       // rotary: pairs sit in adjacent lanes
        float partner = __shfl_xor(val, 1);
        float c, s;
        __sincosf((float)spos * invf, &s, &c);
        val = (lane & 1) ? (val * c + partner * s) : (val * c - partner * s);
        const size_t idx = ((size_t)(bl * NH + h) * SEQ + spos) * DH + f;
        if (p == 0) Qws[idx] = f2bf(val); else Kws[idx] = f2bf(val);
      } else {                           // V stored transposed [head][dh][s]
        Vt[((size_t)(bl * NH + h) * DH + f) * SEQ + spos] = f2bf(val);
      }
    }
  }
}

// ---------- kernel 2: causal flash attention (K-fragments pipelined, 2x unrolled) ----------
#define PSTR 40   // row stride (shorts): 80B -> 16B-aligned rows, conflict-free b128 reads
__global__ __launch_bounds__(256) void attn_kernel(
    const unsigned short* __restrict__ Qws, const unsigned short* __restrict__ Kws,
    const unsigned short* __restrict__ Vt,  unsigned short* __restrict__ inter)
{
  __shared__ unsigned short pbuf[8][16][PSTR];
  const int lane = threadIdx.x & 31;
  const int w    = threadIdx.x >> 5;
  const int hd   = blockIdx.y;                 // bl*8 + h
  const int q0   = blockIdx.x * 128 + w * 16;  // 16 query rows per wave
  const int colL = lane & 15, lh = lane >> 4;

  const unsigned short* Khead = Kws + (size_t)hd * SEQ * DH;
  const unsigned short* Vhead = Vt  + (size_t)hd * DH * SEQ;

  const unsigned short* qrow = Qws + ((size_t)hd * SEQ + q0 + colL) * DH;
  const v16bf aQ0 = ldfragA(qrow, 0, lane);
  const v16bf aQ1 = ldfragA(qrow, 32, lane);

  v8f o[4] = {};
  float mrow[8], lrow[8];
#pragma unroll
  for (int j = 0; j < 8; ++j) { mrow[j] = -__builtin_inff(); lrow[j] = 0.0f; }

  // preload K fragments for block 0
  v16bf kf0, kf1, kf2, kf3;
  {
    const unsigned short* kr = Khead + (size_t)colL * DH;
    kf0 = ldfragB(kr, 0, lane);            kf1 = ldfragB(kr, 32, lane);
    kf2 = ldfragB(kr + 16 * DH, 0, lane);  kf3 = ldfragB(kr + 16 * DH, 32, lane);
  }

  const int nkb = (q0 + 16 + 31) >> 5;
#pragma unroll 2
  for (int kb = 0; kb < nkb; ++kb) {
    const int kbase = kb * 32;

    v8f s0 = {}, s1 = {};
    s0 = wmma_bf16(aQ0, kf0, s0);
    s0 = wmma_bf16(aQ1, kf1, s0);
    s1 = wmma_bf16(aQ0, kf2, s1);
    s1 = wmma_bf16(aQ1, kf3, s1);

    // issue next block's K fragments now; their wait lands after softmax+LDS
    const int knb = (kb + 1 < nkb) ? kbase + 32 : kbase;   // last iter: harmless reload
    v16bf kn0, kn1, kn2, kn3;
    {
      const unsigned short* kr = Khead + (size_t)(knb + colL) * DH;
      kn0 = ldfragB(kr, 0, lane);            kn1 = ldfragB(kr, 32, lane);
      kn2 = ldfragB(kr + 16 * DH, 0, lane);  kn3 = ldfragB(kr + 16 * DH, 32, lane);
    }
    // prefetch next block's V rows (global_prefetch_b8): zero-VGPR warm-up
#pragma unroll
    for (int t = 0; t < 4; ++t)
      __builtin_prefetch(Vhead + (size_t)(t * 16 + colL) * SEQ + knb, 0, 3);

#pragma unroll
    for (int j = 0; j < 8; ++j) {
      const int q = q0 + j + lh * 8;
      float v0 = s0[j] * ATT_SCALE;
      float v1 = s1[j] * ATT_SCALE;
      if (kbase + colL      > q) v0 = -__builtin_inff();
      if (kbase + 16 + colL > q) v1 = -__builtin_inff();
      float mx = fmaxf(v0, v1);                      // row-max over 16 lanes of this half
      mx = fmaxf(mx, __shfl_xor(mx, 1));
      mx = fmaxf(mx, __shfl_xor(mx, 2));
      mx = fmaxf(mx, __shfl_xor(mx, 4));
      mx = fmaxf(mx, __shfl_xor(mx, 8));
      const float mnew  = fmaxf(mrow[j], mx);
      const float alpha = __expf(mrow[j] - mnew);
      const float p0 = __expf(v0 - mnew);
      const float p1 = __expf(v1 - mnew);
      float rs = p0 + p1;
      rs += __shfl_xor(rs, 1);
      rs += __shfl_xor(rs, 2);
      rs += __shfl_xor(rs, 4);
      rs += __shfl_xor(rs, 8);
      lrow[j] = lrow[j] * alpha + rs;
      mrow[j] = mnew;
#pragma unroll
      for (int t = 0; t < 4; ++t) o[t][j] *= alpha;
      pbuf[w][j + lh * 8][colL]      = f2bf(p0);     // C-layout -> LDS
      pbuf[w][j + lh * 8][16 + colL] = f2bf(p1);
    }
    __builtin_amdgcn_wave_barrier();                 // same-wave DS ops are in-order
    const v16bf aP = ldfragA(&pbuf[w][colL][0], 0, lane);   // reload as A-layout
    __builtin_amdgcn_wave_barrier();

#pragma unroll
    for (int t = 0; t < 4; ++t) {
      const unsigned short* vrow = Vhead + (size_t)(t * 16 + colL) * SEQ;
      o[t] = wmma_bf16(aP, ldfragB(vrow, kbase, lane), o[t]);
    }
    kf0 = kn0; kf1 = kn1; kf2 = kn2; kf3 = kn3;
  }

  const int bl = hd >> 3, h = hd & 7;
#pragma unroll
  for (int j = 0; j < 8; ++j) {
    const float inv = 1.0f / lrow[j];
    const int m = (bl << 11) + q0 + j + lh * 8;
    const size_t base = (size_t)m * DIMM + h * DH;
#pragma unroll
    for (int t = 0; t < 4; ++t)
      inter[base + t * 16 + colL] = f2bf(o[t][j] * inv);
  }
}

// ---------- kernel 3: output projection (fp32 out, pipelined, fully unrolled) ----------
__global__ __launch_bounds__(256) void outproj_kernel(
    const unsigned short* __restrict__ interb, const unsigned short* __restrict__ Wob,
    const float* __restrict__ bo, float* __restrict__ out)
{
  const int lane = threadIdx.x & 31;
  const int w  = blockIdx.x * (blockDim.x >> 5) + (threadIdx.x >> 5);
  const int mt = w & 511, ng = w >> 9;
  const int m0 = mt * 16, n0 = ng * 64;
  const int colL = lane & 15, lh = lane >> 4;

  const unsigned short* arow = interb + (size_t)(m0 + colL) * DIMM;
  const unsigned short* wrow[4];
#pragma unroll
  for (int t = 0; t < 4; ++t) wrow[t] = Wob + (size_t)(n0 + t * 16 + colL) * DIMM;

  v16bf a = ldfragA(arow, 0, lane);
  v16bf b[4];
#pragma unroll
  for (int t = 0; t < 4; ++t) b[t] = ldfragB(wrow[t], 0, lane);

  v8f acc[4] = {};
#pragma unroll
  for (int k = 0; k < DIMM; k += 32) {
    const int kn = (k + 32 < DIMM) ? k + 32 : k;
    v16bf an = ldfragA(arow, kn, lane);
    v16bf bn[4];
#pragma unroll
    for (int t = 0; t < 4; ++t) bn[t] = ldfragB(wrow[t], kn, lane);
#pragma unroll
    for (int t = 0; t < 4; ++t) acc[t] = wmma_bf16(a, b[t], acc[t]);
    a = an;
#pragma unroll
    for (int t = 0; t < 4; ++t) b[t] = bn[t];
  }
#pragma unroll
  for (int t = 0; t < 4; ++t) {
    const int col = n0 + t * 16 + colL;
    const float bval = bo[col];
#pragma unroll
    for (int j = 0; j < 8; ++j)
      out[(size_t)(m0 + j + lh * 8) * DIMM + col] = acc[t][j] + bval;
  }
}

// ---------- host ----------
extern "C" void kernel_launch(void* const* d_in, const int* in_sizes, int n_in,
                              void* d_out, int out_size, void* d_ws, size_t ws_size,
                              hipStream_t stream) {
  const float* x  = (const float*)d_in[0];
  const float* Wq = (const float*)d_in[1];
  const float* bq = (const float*)d_in[2];
  const float* Wk = (const float*)d_in[3];
  const float* bk = (const float*)d_in[4];
  const float* Wv = (const float*)d_in[5];
  const float* bv = (const float*)d_in[6];
  const float* Wo = (const float*)d_in[7];
  const float* bo = (const float*)d_in[8];

  unsigned char* ws = (unsigned char*)d_ws;
  size_t off = 0;
  auto alloc = [&](size_t bytes) -> void* {
    void* p = ws + off; off += (bytes + 255) & ~(size_t)255; return p;
  };
  const size_t XB = (size_t)MTOT * DIMM, WB = (size_t)DIMM * DIMM;
  const size_t HB = (size_t)BL * NH * SEQ * DH;
  unsigned short* xb   = (unsigned short*)alloc(XB * 2);
  unsigned short* Wqb  = (unsigned short*)alloc(WB * 2);
  unsigned short* Wkb  = (unsigned short*)alloc(WB * 2);
  unsigned short* Wvb  = (unsigned short*)alloc(WB * 2);
  unsigned short* Wob  = (unsigned short*)alloc(WB * 2);
  unsigned short* Qws  = (unsigned short*)alloc(HB * 2);
  unsigned short* Kws  = (unsigned short*)alloc(HB * 2);
  unsigned short* Vt   = (unsigned short*)alloc(HB * 2);
  unsigned short* intr = (unsigned short*)alloc(XB * 2);
  (void)ws_size; (void)in_sizes; (void)n_in; (void)out_size;

  // 0) convert inputs to bf16
  cvt_f32_bf16_v4<<<(int)(XB / 4 / 256), 256, 0, stream>>>((const float4*)x,  (uint2*)xb,  (int)(XB / 4));
  cvt_f32_bf16_v4<<<(int)(WB / 4 / 256), 256, 0, stream>>>((const float4*)Wq, (uint2*)Wqb, (int)(WB / 4));
  cvt_f32_bf16_v4<<<(int)(WB / 4 / 256), 256, 0, stream>>>((const float4*)Wk, (uint2*)Wkb, (int)(WB / 4));
  cvt_f32_bf16_v4<<<(int)(WB / 4 / 256), 256, 0, stream>>>((const float4*)Wv, (uint2*)Wvb, (int)(WB / 4));
  cvt_f32_bf16_v4<<<(int)(WB / 4 / 256), 256, 0, stream>>>((const float4*)Wo, (uint2*)Wob, (int)(WB / 4));

  // 1) QKV projections + rotary: 3 proj * 8 Ngroups * 512 Mtiles waves, 8 waves/block
  proj_kernel<<<1536, 256, 0, stream>>>(xb, Wqb, Wvb, Wkb, bq, bv, bk, Qws, Kws, Vt);

  // 2) flash attention: 16 q-blocks x 32 heads
  attn_kernel<<<dim3(16, 32), 256, 0, stream>>>(Qws, Kws, Vt, intr);

  // 3) output projection: 8 Ngroups * 512 Mtiles waves
  outproj_kernel<<<512, 256, 0, stream>>>(intr, Wob, bo, (float*)d_out);
}
